// IPNN_21844203668191
// MI455X (gfx1250) — matrix operand order repaired and checked by reading, not compiled, not run on hardware
//
#include <hip/hip_runtime.h>

// ---------------------------------------------------------------------------
// Reference: B=1024 rows, 4 vars x 16 states, J=16^4=65536, Y=100 (pad 128).
//   p0..p3 = softmax over each 16-block of logits
//   flat[b,j] = o01[b,j01]*o23[b,j23],  j = j01*256 + j23
//   prob[y,j] = clip(max(sum_{b in y} flat,eps)/max(sum_b flat,eps),0,1)
//   probs[b,y] = sum_j prob[y,j]*flat[b,j]
// Fused final contraction (exact factorization of flat):
//   probs[b,y] = sum_{j01} o01[b,j01] * (sum_{j23} o23[b,j23]*prob[y,j01*256+j23])
// Inner sum: WMMA bf16 chains (A = o23 resident in VGPRs); outer: fp32 VALU FMA.
// ---------------------------------------------------------------------------
#define BATCH 1024
#define JDIM  65536
#define YPAD  128
#define NCLS  100

typedef __bf16 bf16x16 __attribute__((ext_vector_type(16)));
typedef float  f32x8   __attribute__((ext_vector_type(8)));

union FragBF {
    bf16x16 v;
    uint4   q[2];
};

static __device__ __forceinline__ unsigned short f2bf(float f) {
    union { float f; unsigned u; } x; x.f = f;
    unsigned r = x.u + 0x7FFFu + ((x.u >> 16) & 1u);  // round-to-nearest-even
    return (unsigned short)(r >> 16);
}

// ---------------------------------------------------------------------------
// Kernel 0: zero the output (split-j01 GEMM accumulates with f32 atomics).
// ---------------------------------------------------------------------------
__global__ void zero_out_kernel(float* __restrict__ p, int n) {
    int i = blockIdx.x * 256 + threadIdx.x;
    if (i < n) p[i] = 0.0f;
}

// ---------------------------------------------------------------------------
// Kernel 1: per-row softmaxes, pairwise outer products, label extract.
// One block (256 threads) per batch row. Emits o01 (f32), o23 (f32 + bf16).
// ---------------------------------------------------------------------------
__global__ void __launch_bounds__(256)
softmax_outer_kernel(const float* __restrict__ logits,
                     const float* __restrict__ y_true,
                     float* __restrict__ o01,
                     float* __restrict__ o23,
                     unsigned short* __restrict__ o23bf,
                     int*   __restrict__ label) {
    const int b = blockIdx.x;
    const int t = threadIdx.x;
    __shared__ float sl[64], se[64], sp[64], gm[4], gs[4];

    if (t < 64) sl[t] = logits[b * 64 + t];
    __syncthreads();
    if (t < 4) {
        float m = sl[t * 16];
        #pragma unroll
        for (int i = 1; i < 16; ++i) m = fmaxf(m, sl[t * 16 + i]);
        gm[t] = m;
    }
    __syncthreads();
    if (t < 64) se[t] = __expf(sl[t] - gm[t >> 4]);
    __syncthreads();
    if (t < 4) {
        float s = 0.f;
        #pragma unroll
        for (int i = 0; i < 16; ++i) s += se[t * 16 + i];
        gs[t] = s;
    }
    __syncthreads();
    if (t < 64) sp[t] = se[t] / gs[t >> 4];
    __syncthreads();

    float a01 = sp[t >> 4]        * sp[16 + (t & 15)];
    float a23 = sp[32 + (t >> 4)] * sp[48 + (t & 15)];
    o01[b * 256 + t]   = a01;
    o23[b * 256 + t]   = a23;
    o23bf[b * 256 + t] = f2bf(a23);

    if (t < NCLS && y_true[b * NCLS + t] > 0.5f) label[b] = t;
}

// ---------------------------------------------------------------------------
// Kernel 2: class-conditional sums + clamp -> probT (bf16).
// Grid: 512 blocks x 128 threads; block owns a 128-wide j strip.
// LDS bins[100][128] f32 = 51.2 KB; lane t owns column t -> no atomics/syncs.
// probT is [YPAD=128][JDIM] row-major (class = row); rows 100..127 zeroed.
// ---------------------------------------------------------------------------
__global__ void __launch_bounds__(128)
build_prob_kernel(const float* __restrict__ o01,
                  const float* __restrict__ o23,
                  const int*   __restrict__ label,
                  unsigned short* __restrict__ probT) {
    const int j01  = blockIdx.x >> 1;                 // 0..255
    const int j23b = (blockIdx.x & 1) * 128;          // 0 or 128
    const int t    = threadIdx.x;                     // 0..127
    const int j23  = j23b + t;

    __shared__ float bins[NCLS * 128];
    #pragma unroll 4
    for (int y = 0; y < NCLS; ++y) bins[y * 128 + t] = 0.f;

    const size_t col = (size_t)j01 * 256 + j23;

    for (int b = 0; b < BATCH; ++b) {
        float a = o01[b * 256 + j01];                 // uniform -> scalar load
        float c = o23[b * 256 + j23];                 // coalesced
        float v = a * c;
        int y = label[b];
        y = y < 0 ? 0 : (y > NCLS - 1 ? NCLS - 1 : y);
        bins[y * 128 + t] += v;                       // column-exclusive
    }

    float den = 0.f;
    #pragma unroll 4
    for (int y = 0; y < NCLS; ++y) den += bins[y * 128 + t];
    den = fmaxf(den, 1e-6f);

    #pragma unroll 4
    for (int y = 0; y < NCLS; ++y) {
        float num = fmaxf(bins[y * 128 + t], 1e-6f);
        float p = num / den;
        p = p > 1.f ? 1.f : p;
        probT[(size_t)y * JDIM + col] = f2bf(p);
    }
    for (int y = NCLS; y < YPAD; ++y)
        probT[(size_t)y * JDIM + col] = 0;
}

// ---------------------------------------------------------------------------
// Kernel 3 (fused bilinear):
//   probs[b,y] += o01[b,j01] * T[b,y],
//   T[b,y] = sum_{j23<256} o23bf[b,j23] * probT[y, j01*256+j23]  (WMMA, K=256)
//
// Block = 256 threads = 8 wave32. Wave w: M rows m0 = bx*64+(w&3)*16,
// N cols n0 = (w>>2)*64 (4 tiles of 16). Grid: (16 M-tiles, 16 j01-chunks);
// fp32 atomic writeout.
//
// A (o23bf rows) is loaded ONCE into 8 resident fragments (K=0..255) and
// reused for all 16 j01 steps x 4 N-tiles = 512 WMMAs per wave.
//
// ILP: kc loop OUTER, 4 N-tiles run 4 independent accumulator chains per
// k-step. All B addresses derive from ONE per-step base pointer with
// compile-time element offsets nt*16*JDIM + kc*32 (byte offsets <= 6 MB+448,
// inside the signed 24-bit IOFFSET) -> single-VADDR global_load_b128 clauses,
// not FLAT (avoids the DScnt-coupled flat path).
//
// Fragment layouts used here (wave32):
//  A 16x32 bf16: lane L: M=L&15, h=L>>4; VGPR0..3: K=h*16..h*16+7,
//                VGPR4..7: K=h*16+8..h*16+15          -> two 16B loads.
//  B 32x16 bf16 (probT row = class): lane L: N=L&15, h=L>>4;
//                K = h*16 .. h*16+15 contiguous         -> two 16B loads.
//  D 16x16 f32:  element v, lane L: M = m0+v+8*h, N = n0+(L&15).
// ---------------------------------------------------------------------------
__global__ void __launch_bounds__(256)
fused_bilinear_kernel(const unsigned short* __restrict__ o23bf,
                      const float* __restrict__ o01,
                      const unsigned short* __restrict__ probT,
                      float* __restrict__ out) {
    const int lane = threadIdx.x & 31;
    const int wave = threadIdx.x >> 5;            // 0..7
    const int m0   = blockIdx.x * 64 + (wave & 3) * 16;
    const int n0   = (wave >> 2) * 64;
    const int j01b = blockIdx.y * 16;             // this block's j01 chunk
    const int l15  = lane & 15;
    const int h    = lane >> 4;

    // Resident A fragments: o23bf[m0+l15][0..255]
    FragBF afrag[8];
    {
        const unsigned short* aRow = o23bf + (size_t)(m0 + l15) * 256;
        #pragma unroll
        for (int kc = 0; kc < 8; ++kc) {
            afrag[kc].q[0] = *(const uint4*)(aRow + kc * 32 + h * 16);
            afrag[kc].q[1] = *(const uint4*)(aRow + kc * 32 + h * 16 + 8);
        }
    }

    f32x8 acc[4];
    #pragma unroll
    for (int i = 0; i < 4; ++i)
        #pragma unroll
        for (int e = 0; e < 8; ++e) acc[i][e] = 0.f;

    for (int s = 0; s < 16; ++s) {
        const int j01 = j01b + s;

        // Single base for all 4 N-tiles this step; nt/kc deltas are constants.
        const unsigned short* bRow =
            probT + (size_t)(n0 + l15) * JDIM + (size_t)j01 * 256 + h * 16;
        __builtin_prefetch(bRow + 256, 0, 1);   // next j01, global_prefetch_b8
        __builtin_prefetch(bRow + (size_t)32 * JDIM + 256, 0, 1);

        // Row scales o01[m0+v+8h, j01] (uniform within each 16-lane half).
        float w01[8];
        #pragma unroll
        for (int v = 0; v < 8; ++v)
            w01[v] = o01[(size_t)(m0 + v + 8 * h) * 256 + j01];

        f32x8 tacc[4];
        #pragma unroll
        for (int nt = 0; nt < 4; ++nt)
            #pragma unroll
            for (int e = 0; e < 8; ++e) tacc[nt][e] = 0.f;

        #pragma unroll
        for (int kc = 0; kc < 8; ++kc) {
            FragBF bfrag[4];
            #pragma unroll
            for (int nt = 0; nt < 4; ++nt) {
                bfrag[nt].q[0] =
                    *(const uint4*)(bRow + (size_t)nt * 16 * JDIM + kc * 32);
                bfrag[nt].q[1] =
                    *(const uint4*)(bRow + (size_t)nt * 16 * JDIM + kc * 32 + 8);
            }
            #pragma unroll
            for (int nt = 0; nt < 4; ++nt) {
                tacc[nt] = __builtin_amdgcn_wmma_f32_16x16x32_bf16(
                    /*neg_a=*/false, afrag[kc].v, /*neg_b=*/false, bfrag[nt].v,
                    /*c_mod=*/(short)0, tacc[nt],
                    /*reuse_a=*/false, /*reuse_b=*/false);
            }
        }

        #pragma unroll
        for (int nt = 0; nt < 4; ++nt)
            #pragma unroll
            for (int v = 0; v < 8; ++v)
                acc[nt][v] += w01[v] * tacc[nt][v];
    }

    #pragma unroll
    for (int nt = 0; nt < 4; ++nt) {
        int n = n0 + nt * 16 + l15;
        if (n < NCLS) {
            #pragma unroll
            for (int v = 0; v < 8; ++v) {
                int m = m0 + v + 8 * h;
                atomicAdd(out + m * NCLS + n, acc[nt][v]);
            }
        }
    }
}

// ---------------------------------------------------------------------------
// Launch. Workspace layout (bytes), total ~= 20 MB:
//   [0, 1MB)             o01    1024x256 f32
//   [1MB, 2MB)           o23    1024x256 f32
//   [2MB, 2.5MB)         o23bf  1024x256 bf16
//   [2.5MB, 2.5MB+4KB)   label  1024 i32
//   [3MB, 3MB+16.8MB)    probT  128x65536 bf16
// ---------------------------------------------------------------------------
extern "C" void kernel_launch(void* const* d_in, const int* in_sizes, int n_in,
                              void* d_out, int out_size, void* d_ws, size_t ws_size,
                              hipStream_t stream) {
    const float* logits = (const float*)d_in[0];   // [1024, 64]
    const float* y_true = (const float*)d_in[1];   // [1024, 100]
    float* out = (float*)d_out;                    // [1024, 100]

    char* ws = (char*)d_ws;
    float*          o01   = (float*)(ws);
    float*          o23   = (float*)(ws + ((size_t)1 << 20));
    unsigned short* o23bf = (unsigned short*)(ws + ((size_t)2 << 20));
    int*            lab   = (int*)(ws + ((size_t)5 << 19));            // 2.5 MB
    unsigned short* probT = (unsigned short*)(ws + ((size_t)3 << 20)); // 3 MB

    zero_out_kernel<<<(BATCH * NCLS + 255) / 256, 256, 0, stream>>>(out, BATCH * NCLS);
    softmax_outer_kernel<<<BATCH, 256, 0, stream>>>(logits, y_true, o01, o23, o23bf, lab);
    build_prob_kernel<<<512, 128, 0, stream>>>(o01, o23, lab, probT);
    fused_bilinear_kernel<<<dim3(BATCH / 64, JDIM / 256 / 16), 256, 0, stream>>>(
        o23bf, o01, probT, out);
}
